// HybridVisionNystromformer_72499047956691
// MI455X (gfx1250) — matrix operand order repaired and checked by reading, not compiled, not run on hardware
//
#include <hip/hip_runtime.h>
#include <hip/hip_bf16.h>

// ---------------------------------------------------------------------------
// MI455X (gfx1250) HybridVisionNystromformer pipeline, wave32 + WMMA.
// GEMM-shaped ops -> v_wmma_f32_16x16x32_f16 ; pinv -> v_wmma_f32_16x16x4_f32.
// LDS tiles are [outer][k] pitch 40 halves (80B) -> fragments are ds_load_b128.
// Contiguous staging uses GLOBAL_LOAD_ASYNC_TO_LDS_B128 (ASYNCcnt) via the
// gfx1250 builtin (signature: (global int4*, local int4*, imm off, imm cpol)).
// ---------------------------------------------------------------------------

typedef __attribute__((ext_vector_type(16))) _Float16 v16h;
typedef __attribute__((ext_vector_type(8)))  _Float16 v8h;
typedef __attribute__((ext_vector_type(8)))  float    v8f;
typedef __attribute__((ext_vector_type(2)))  float    v2f;
typedef __attribute__((ext_vector_type(4)))  int      v4i;

#define DEV __device__ __forceinline__

constexpr int LDA = 40;  // LDS pitch in halves (80B, multiple of 16B)

DEV int clampi(int v, int lo, int hi) { return v < lo ? lo : (v > hi ? hi : v); }

// ---------------- async global->LDS staging (gfx1250 path) ----------------
#if defined(__has_builtin)
#if __has_builtin(__builtin_amdgcn_global_load_async_to_lds_b128)
#define USE_ASYNC_LDS 1
#endif
#endif

#ifdef USE_ASYNC_LDS
typedef __attribute__((address_space(1))) v4i g_v4i;  // global int4
typedef __attribute__((address_space(3))) v4i l_v4i;  // LDS int4
DEV void stage_b128(_Float16* lds, const _Float16* g) {
  __builtin_amdgcn_global_load_async_to_lds_b128((g_v4i*)g, (l_v4i*)lds, 0, 0);
}
DEV void stage_wait() {
#if __has_builtin(__builtin_amdgcn_s_wait_asynccnt)
  __builtin_amdgcn_s_wait_asynccnt(0);
#else
  asm volatile("s_wait_asynccnt 0x0" ::: "memory");
#endif
}
#else
DEV void stage_b128(_Float16* lds, const _Float16* g) {
  *(v8h*)lds = *(const v8h*)g;
}
DEV void stage_wait() {}
#endif

// ---------------- WMMA wrappers ----------------
DEV v8f wmma16(v16h a, v16h b, v8f c) {
  return __builtin_amdgcn_wmma_f32_16x16x32_f16(false, a, false, b, (short)0, c,
                                                false, false);
}
DEV v8f wmma4(v2f a, v2f b, v8f c) {
  return __builtin_amdgcn_wmma_f32_16x16x4_f32(false, a, false, b, (short)0, c,
                                               false, false);
}

// A 16x32 (MxK) f16: lane L -> M = L&15; K = 16g + e
DEV v16h frag_a16(const _Float16* As, int m_base, int lane) {
  const int g = lane >> 4;
  const _Float16* row = As + (m_base + (lane & 15)) * LDA;
  const v8h lo = *(const v8h*)(row + 16 * g);
  const v8h hi = *(const v8h*)(row + 16 * g + 8);
  return __builtin_shufflevector(lo, hi, 0, 1, 2, 3, 4, 5, 6, 7, 8, 9, 10, 11,
                                 12, 13, 14, 15);
}
// B 32x16 (KxN) f16 from N-major LDS tile: lane L -> N = L&15; K = 16g + e
DEV v16h frag_b16(const _Float16* BsT, int n_base, int lane) {
  const int g = lane >> 4;
  const _Float16* row = BsT + (n_base + (lane & 15)) * LDA + 16 * g;
  const v8h lo = *(const v8h*)(row);
  const v8h hi = *(const v8h*)(row + 8);
  return __builtin_shufflevector(lo, hi, 0, 1, 2, 3, 4, 5, 6, 7, 8, 9, 10, 11,
                                 12, 13, 14, 15);
}

DEV void gemm_step(const _Float16* As, const _Float16* BsT, int m_sub,
                   int n_sub0, int lane, v8f& acc0, v8f& acc1) {
  v16h a = frag_a16(As, m_sub * 16, lane);
  v16h b0 = frag_b16(BsT, n_sub0 * 16, lane);
  v16h b1 = frag_b16(BsT, (n_sub0 + 1) * 16, lane);
  acc0 = wmma16(a, b0, acc0);
  acc1 = wmma16(a, b1, acc1);
}

// C/D layout: element r -> M = r + 8g, N = lane&15
DEV void store_c_lds(float* Cs, int m_sub, int n_sub0, int lane, const v8f& acc0,
                     const v8f& acc1) {
  const int g = lane >> 4, l = lane & 15;
#pragma unroll
  for (int r = 0; r < 8; ++r) {
    const int m = m_sub * 16 + r + 8 * g;
    Cs[m * 64 + n_sub0 * 16 + l] = acc0[r];
    Cs[m * 64 + (n_sub0 + 1) * 16 + l] = acc1[r];
  }
}

// ---------------- weight pre-convert / pre-transpose -----------------------
__global__ __launch_bounds__(256) void cvt_f16_kernel(
    const float* __restrict__ src, _Float16* __restrict__ dst, int n) {
  const int i = blockIdx.x * 256 + threadIdx.x;
  if (i < n) dst[i] = (_Float16)src[i];
}
// src[R][C] (f32) -> dst[C][R] (f16)
__global__ __launch_bounds__(256) void transpose_f16_kernel(
    const float* __restrict__ src, _Float16* __restrict__ dst, int R, int C) {
  const int i = blockIdx.x * 256 + threadIdx.x;
  if (i < R * C) {
    const int r = i / C, c = i % C;
    dst[(size_t)c * R + r] = (_Float16)src[i];
  }
}

// ===========================================================================
// conv0 (1->64, 3x3, pad1) + BN + ReLU, direct VALU, branch-free
// ===========================================================================
__global__ __launch_bounds__(256) void conv0_kernel(
    const float* __restrict__ x, const float* __restrict__ w,
    const float* __restrict__ bias, const float* __restrict__ g,
    const float* __restrict__ be, const float* __restrict__ mu,
    const float* __restrict__ var, _Float16* __restrict__ out) {
  const int idx = blockIdx.x * 256 + threadIdx.x;  // 4*64*256*256
  const int b = idx >> 22;
  const int c = (idx >> 16) & 63;
  const int y = (idx >> 8) & 255;
  const int xx = idx & 255;
  float acc = 0.f;
#pragma unroll
  for (int r = 0; r < 3; ++r)
#pragma unroll
    for (int s = 0; s < 3; ++s) {
      const int iy = y + r - 1, ix = xx + s - 1;
      const int cy = clampi(iy, 0, 255), cx = clampi(ix, 0, 255);
      const float v = x[(size_t)b * 65536 + cy * 256 + cx];
      acc += ((iy == cy) && (ix == cx)) ? v * w[c * 9 + r * 3 + s] : 0.f;
    }
  const float sc = g[c] * rsqrtf(var[c] + 1e-5f);
  const float v = (acc + bias[c]) * sc + (be[c] - mu[c] * sc);
  out[((size_t)(b * 64 + c) << 16) + y * 256 + xx] = (_Float16)fmaxf(v, 0.f);
}

// ===========================================================================
// conv 3x3 s1 p1 implicit GEMM + BN + ReLU.  grid = (HW/64, COUT/64, B)
// ===========================================================================
template <int CIN, int COUT>
__global__ __launch_bounds__(256) void conv3x3_wmma(
    const _Float16* __restrict__ in, const _Float16* __restrict__ wf,
    const float* __restrict__ bias, const float* __restrict__ g,
    const float* __restrict__ be, const float* __restrict__ mu,
    const float* __restrict__ var, _Float16* __restrict__ out) {
  constexpr int KC = CIN * 9;
  __shared__ __align__(16) _Float16 As[64 * LDA];
  __shared__ __align__(16) _Float16 Bs[64 * LDA];
  const int tid = threadIdx.x, wave = tid >> 5, lane = tid & 31;
  const int m_sub = wave & 3, n_sub0 = (wave >> 2) << 1;
  const int pix0 = blockIdx.x * 64;
  const int cout0 = blockIdx.y * 64;
  const int b = blockIdx.z;
  const int sm = tid >> 2, skk = (tid & 3) * 8;  // staging coords
  const int sp = pix0 + sm, sy = sp >> 8, sx = sp & 255;
  v8f acc0 = {}, acc1 = {};
  for (int k0 = 0; k0 < KC; k0 += 32) {
    __syncthreads();
    stage_b128(As + sm * LDA + skk,
               wf + (size_t)(cout0 + sm) * KC + k0 + skk);  // weights (async)
    v8h pack;
#pragma unroll
    for (int e = 0; e < 8; ++e) {  // im2col gather (register pack)
      const int k = k0 + skk + e;
      const int ci = k / 9, rs = k % 9, r = rs / 3, s = rs % 3;
      const int iy = sy + r - 1, ix = sx + s - 1;
      const int cy = clampi(iy, 0, 255), cx = clampi(ix, 0, 255);
      const _Float16 v = in[(((size_t)b * CIN + ci) << 16) + cy * 256 + cx];
      pack[e] = ((iy == cy) && (ix == cx)) ? v : (_Float16)0;
    }
    *(v8h*)(Bs + sm * LDA + skk) = pack;
    stage_wait();
    __syncthreads();
    gemm_step(As, Bs, m_sub, n_sub0, lane, acc0, acc1);
  }
  const int gi = lane >> 4, l15 = lane & 15;
#pragma unroll
  for (int r = 0; r < 8; ++r) {
    const int m = cout0 + m_sub * 16 + r + 8 * gi;
    const float sc = g[m] * rsqrtf(var[m] + 1e-5f);
    const float sh = be[m] + (bias[m] - mu[m]) * sc;
#pragma unroll
    for (int t = 0; t < 2; ++t) {
      const int p = pix0 + (n_sub0 + t) * 16 + l15;
      const float v = (t ? acc1[r] : acc0[r]) * sc + sh;
      out[(((size_t)b * COUT + m) << 16) + p] = (_Float16)fmaxf(v, 0.f);
    }
  }
}

// ===========================================================================
// conv3: 256->800, 4x4 s4 p0 implicit GEMM (K=4096) -> tokens [B][4096][800]
// grid = (4096/64, 13, B)
// ===========================================================================
__global__ __launch_bounds__(256) void conv4x4_wmma(
    const _Float16* __restrict__ in, const _Float16* __restrict__ wf,
    const float* __restrict__ bias, const float* __restrict__ g,
    const float* __restrict__ be, const float* __restrict__ mu,
    const float* __restrict__ var, _Float16* __restrict__ tok) {
  __shared__ __align__(16) _Float16 As[64 * LDA];
  __shared__ __align__(16) _Float16 Bs[64 * LDA];
  const int tid = threadIdx.x, wave = tid >> 5, lane = tid & 31;
  const int m_sub = wave & 3, n_sub0 = (wave >> 2) << 1;
  const int pix0 = blockIdx.x * 64;
  const int cout0 = blockIdx.y * 64;  // rows >= 800 masked
  const int b = blockIdx.z;
  const int sm = tid >> 2, skk = (tid & 3) * 8;
  const int sp = pix0 + sm, sty = sp >> 6, stx = sp & 63;
  const bool arow_ok = (cout0 + sm) < 800;
  if (!arow_ok) {  // zero masked A rows once (never overwritten)
    v8h z;
#pragma unroll
    for (int e = 0; e < 8; ++e) z[e] = (_Float16)0;
    *(v8h*)(As + sm * LDA + skk) = z;
  }
  v8f acc0 = {}, acc1 = {};
  for (int k0 = 0; k0 < 4096; k0 += 32) {
    __syncthreads();
    if (arow_ok)
      stage_b128(As + sm * LDA + skk,
                 wf + (size_t)(cout0 + sm) * 4096 + k0 + skk);
    v8h pack;
#pragma unroll
    for (int e = 0; e < 8; ++e) {
      const int k = k0 + skk + e;
      const int ci = k >> 4, rs = k & 15, r = rs >> 2, s = rs & 3;
      pack[e] = in[(((size_t)b * 256 + ci) << 16) + (sty * 4 + r) * 256 +
                   (stx * 4 + s)];
    }
    *(v8h*)(Bs + sm * LDA + skk) = pack;
    stage_wait();
    __syncthreads();
    gemm_step(As, Bs, m_sub, n_sub0, lane, acc0, acc1);
  }
  const int gi = lane >> 4, l15 = lane & 15;
#pragma unroll
  for (int r = 0; r < 8; ++r) {
    const int m = cout0 + m_sub * 16 + r + 8 * gi;
    if (m < 800) {
      const float sc = g[m] * rsqrtf(var[m] + 1e-5f);
      const float sh = be[m] + (bias[m] - mu[m]) * sc;
#pragma unroll
      for (int t = 0; t < 2; ++t) {
        const int p = pix0 + (n_sub0 + t) * 16 + l15;
        const float v = (t ? acc1[r] : acc0[r]) * sc + sh;
        tok[((size_t)b * 4096 + p) * 800 + m] = (_Float16)fmaxf(v, 0.f);
      }
    }
  }
}

// ===========================================================================
// qkv: [16384 x 800] @ [800 x 1536] -> Q,K ([BH][N][DH]) and V^T ([BH][DH][N])
// grid = (256, 24)
// ===========================================================================
__global__ __launch_bounds__(256) void qkv_wmma(
    const _Float16* __restrict__ tok, const _Float16* __restrict__ WqkvT,
    _Float16* __restrict__ Qb, _Float16* __restrict__ Kb,
    _Float16* __restrict__ VT) {
  __shared__ __align__(16) _Float16 As[64 * LDA];
  __shared__ __align__(16) _Float16 Bs[64 * LDA];
  const int tid = threadIdx.x, wave = tid >> 5, lane = tid & 31;
  const int m_sub = wave & 3, n_sub0 = (wave >> 2) << 1;
  const int row0 = blockIdx.x * 64;
  const int col0 = blockIdx.y * 64;
  const int sm = tid >> 2, skk = (tid & 3) * 8;
  v8f acc0 = {}, acc1 = {};
  for (int k0 = 0; k0 < 800; k0 += 32) {
    __syncthreads();
    stage_b128(As + sm * LDA + skk,
               tok + (size_t)(row0 + sm) * 800 + k0 + skk);
    stage_b128(Bs + sm * LDA + skk,
               WqkvT + (size_t)(col0 + sm) * 800 + k0 + skk);
    if (tid == 0 && k0 + 32 < 800)
      __builtin_prefetch(WqkvT + (size_t)col0 * 800 + k0 + 32, 0, 0);
    stage_wait();
    __syncthreads();
    gemm_step(As, Bs, m_sub, n_sub0, lane, acc0, acc1);
  }
  const int gi = lane >> 4, l15 = lane & 15;
#pragma unroll
  for (int t = 0; t < 2; ++t) {
    const int col = col0 + (n_sub0 + t) * 16 + l15;
    const int part = col >> 9;  // 0=q 1=k 2=v
    const int wi = col & 511, h = wi >> 6, d = wi & 63;
    const float scale = (part == 0) ? 0.125f : 1.0f;  // DH^-0.5 on q
#pragma unroll
    for (int r = 0; r < 8; ++r) {
      const int trow = row0 + m_sub * 16 + r + 8 * gi;
      const int b = trow >> 12, tk = trow & 4095;
      const _Float16 v = (_Float16)((t ? acc1[r] : acc0[r]) * scale);
      if (part == 2)
        VT[((size_t)(b * 8 + h) * 64 + d) * 4096 + tk] = v;
      else
        (part == 0 ? Qb : Kb)[((size_t)(b * 8 + h) * 4096 + tk) * 64 + d] = v;
    }
  }
}

// landmark means over 64-token blocks: [BH][4096][64] -> [BH][64][64]
__global__ __launch_bounds__(256) void landmark_kernel(
    const _Float16* __restrict__ src, _Float16* __restrict__ dst) {
  const int idx = blockIdx.x * 256 + threadIdx.x;  // 32*64*64
  const int bh = idx >> 12, rem = idx & 4095, m = rem >> 6, d = rem & 63;
  float s = 0.f;
  for (int l = 0; l < 64; ++l)
    s += (float)src[((size_t)bh * 4096 + m * 64 + l) * 64 + d];
  dst[((size_t)bh * 64 + m) * 64 + d] = (_Float16)(s * 0.015625f);
}

// ===========================================================================
// a1 = softmax(Q @ KL^T)  (IS_A2=false, f16 out, grid (64,32))
// a2 = softmax(QL @ KL^T) (IS_A2=true,  f32 out, grid (1,32))
// ===========================================================================
template <bool IS_A2>
__global__ __launch_bounds__(256) void attn_small_wmma(
    const _Float16* __restrict__ Amat, const _Float16* __restrict__ KL,
    _Float16* __restrict__ out16, float* __restrict__ out32) {
  __shared__ __align__(16) _Float16 As[64 * LDA];
  __shared__ __align__(16) _Float16 Bs[64 * LDA];
  __shared__ float Cs[64 * 64];
  const int tid = threadIdx.x, wave = tid >> 5, lane = tid & 31;
  const int m_sub = wave & 3, n_sub0 = (wave >> 2) << 1;
  const int rowsPerBH = IS_A2 ? 64 : 4096;
  const int row0 = blockIdx.x * 64;
  const int bh = blockIdx.y;
  const int sm = tid >> 2, skk = (tid & 3) * 8;
  v8f acc0 = {}, acc1 = {};
  for (int k0 = 0; k0 < 64; k0 += 32) {
    __syncthreads();
    stage_b128(As + sm * LDA + skk,
               Amat + ((size_t)bh * rowsPerBH + row0 + sm) * 64 + k0 + skk);
    stage_b128(Bs + sm * LDA + skk,
               KL + ((size_t)bh * 64 + sm) * 64 + k0 + skk);
    stage_wait();
    __syncthreads();
    gemm_step(As, Bs, m_sub, n_sub0, lane, acc0, acc1);
  }
  store_c_lds(Cs, m_sub, n_sub0, lane, acc0, acc1);
  __syncthreads();
  if (tid < 64) {  // full row softmax (length 64)
    const int row = tid;
    float mx = -1e30f;
    for (int j = 0; j < 64; ++j) mx = fmaxf(mx, Cs[row * 64 + j]);
    float sum = 0.f;
    for (int j = 0; j < 64; ++j) {
      const float e = __expf(Cs[row * 64 + j] - mx);
      Cs[row * 64 + j] = e;
      sum += e;
    }
    const float inv = 1.f / sum;
    for (int j = 0; j < 64; ++j) {
      const float p = Cs[row * 64 + j] * inv;
      if (IS_A2)
        out32[(size_t)bh * 4096 + row * 64 + j] = p;
      else
        out16[((size_t)bh * 4096 + row0 + row) * 64 + j] = (_Float16)p;
    }
  }
}

// ===========================================================================
// s3: S = QL @ K^T [64 x 4096]; store exp(S) f16 + per-row sums via atomics.
// grid = (64, 32)
// ===========================================================================
__global__ __launch_bounds__(256) void s3_wmma(
    const _Float16* __restrict__ QL, const _Float16* __restrict__ Kb,
    _Float16* __restrict__ S3E, float* __restrict__ RS3) {
  __shared__ __align__(16) _Float16 As[64 * LDA];
  __shared__ __align__(16) _Float16 Bs[64 * LDA];
  __shared__ float Cs[64 * 64];
  const int tid = threadIdx.x, wave = tid >> 5, lane = tid & 31;
  const int m_sub = wave & 3, n_sub0 = (wave >> 2) << 1;
  const int tok0 = blockIdx.x * 64;
  const int bh = blockIdx.y;
  const int sm = tid >> 2, skk = (tid & 3) * 8;
  v8f acc0 = {}, acc1 = {};
  for (int k0 = 0; k0 < 64; k0 += 32) {
    __syncthreads();
    stage_b128(As + sm * LDA + skk,
               QL + ((size_t)bh * 64 + sm) * 64 + k0 + skk);
    stage_b128(Bs + sm * LDA + skk,
               Kb + ((size_t)bh * 4096 + tok0 + sm) * 64 + k0 + skk);
    stage_wait();
    __syncthreads();
    gemm_step(As, Bs, m_sub, n_sub0, lane, acc0, acc1);
  }
  store_c_lds(Cs, m_sub, n_sub0, lane, acc0, acc1);
  __syncthreads();
  for (int i = tid; i < 4096; i += 256) {
    const float e = __expf(Cs[i]);
    Cs[i] = e;
    S3E[((size_t)bh * 64 + (i >> 6)) * 4096 + tok0 + (i & 63)] = (_Float16)e;
  }
  __syncthreads();
  if (tid < 64) {
    float s = 0.f;
    for (int j = 0; j < 64; ++j) s += Cs[tid * 64 + j];
    atomicAdd(&RS3[bh * 64 + tid], s);
  }
}

// ===========================================================================
// a3v^T = ((exp(S) @ V) / rowsum)^T -> A3VT [BH][DH][M] f16.  grid = (32)
// ===========================================================================
__global__ __launch_bounds__(256) void a3v_wmma(
    const _Float16* __restrict__ S3E, const _Float16* __restrict__ VT,
    const float* __restrict__ RS3, _Float16* __restrict__ A3VT) {
  __shared__ __align__(16) _Float16 As[64 * LDA];
  __shared__ __align__(16) _Float16 Bs[64 * LDA];
  const int tid = threadIdx.x, wave = tid >> 5, lane = tid & 31;
  const int m_sub = wave & 3, n_sub0 = (wave >> 2) << 1;
  const int bh = blockIdx.x;
  const int sm = tid >> 2, skk = (tid & 3) * 8;
  v8f acc0 = {}, acc1 = {};
  for (int k0 = 0; k0 < 4096; k0 += 32) {
    __syncthreads();
    stage_b128(As + sm * LDA + skk,
               S3E + ((size_t)bh * 64 + sm) * 4096 + k0 + skk);
    stage_b128(Bs + sm * LDA + skk,
               VT + ((size_t)bh * 64 + sm) * 4096 + k0 + skk);
    if (tid == 0 && k0 + 32 < 4096)
      __builtin_prefetch(VT + (size_t)bh * 64 * 4096 + k0 + 32, 0, 0);
    stage_wait();
    __syncthreads();
    gemm_step(As, Bs, m_sub, n_sub0, lane, acc0, acc1);
  }
  const int gi = lane >> 4, l15 = lane & 15;
#pragma unroll
  for (int r = 0; r < 8; ++r) {
    const int m = m_sub * 16 + r + 8 * gi;  // landmark
    const float inv = 1.f / RS3[bh * 64 + m];
#pragma unroll
    for (int t = 0; t < 2; ++t) {
      const int n = (n_sub0 + t) * 16 + l15;  // DH
      A3VT[((size_t)bh * 64 + n) * 64 + m] =
          (_Float16)((t ? acc1[r] : acc0[r]) * inv);
    }
  }
}

// ===========================================================================
// pinv: per (b,h) 6 Newton iterations on 64x64 in LDS, f32 WMMA 16x16x4.
// z = 0.25 z (13I - az(15I - az(7I - az)))      grid = (32)
// ===========================================================================
template <int BMODE>  // 0 = plain B ; else B -> (BMODE*I - B)
DEV void matmul64_f32(float* D, const float* A, const float* B, float scale,
                      int tid) {
  const int wave = tid >> 5, lane = tid & 31;
  const int m_sub = wave & 3, n_sub0 = (wave >> 2) << 1;
  const int g = lane >> 4, l15 = lane & 15;
  v8f acc0 = {}, acc1 = {};
  for (int kc = 0; kc < 16; ++kc) {
    const int kb = kc * 4 + 2 * g;  // A 16x4: lanes>=16 hold K=2,3
    v2f a = *(const v2f*)(A + (m_sub * 16 + l15) * 64 + kb);
#pragma unroll
    for (int t = 0; t < 2; ++t) {
      const int n = (n_sub0 + t) * 16 + l15;
      v2f bf;
#pragma unroll
      for (int e = 0; e < 2; ++e) {
        const int kk = kb + e;
        const float raw = B[kk * 64 + n];
        bf[e] = (BMODE == 0) ? raw
                             : (((kk == n) ? (float)BMODE : 0.0f) - raw);
      }
      if (t == 0)
        acc0 = wmma4(a, bf, acc0);
      else
        acc1 = wmma4(a, bf, acc1);
    }
  }
  __syncthreads();  // all reads done before any aliased write
#pragma unroll
  for (int r = 0; r < 8; ++r) {
    const int m = m_sub * 16 + r + 8 * g;
    D[m * 64 + n_sub0 * 16 + l15] = acc0[r] * scale;
    D[m * 64 + (n_sub0 + 1) * 16 + l15] = acc1[r] * scale;
  }
  __syncthreads();
}

__global__ __launch_bounds__(256) void pinv_kernel(
    const float* __restrict__ A2S, _Float16* __restrict__ ZT) {
  __shared__ float Zs[64 * 64];
  __shared__ float T1[64 * 64];
  __shared__ float T2[64 * 64];
  __shared__ float red[130];
  const int tid = threadIdx.x;
  const int bh = blockIdx.x;
  const float* Ag = A2S + (size_t)bh * 4096;
  for (int i = tid; i < 130; i += 256) red[i] = 0.f;
  __syncthreads();
  for (int i = tid; i < 4096; i += 256) {
    const float v = fabsf(Ag[i]);
    atomicAdd(&red[i >> 6], v);         // row sums
    atomicAdd(&red[64 + (i & 63)], v);  // col sums
  }
  __syncthreads();
  if (tid == 0) {
    float mr = 0.f, mc = 0.f;
    for (int i = 0; i < 64; ++i) {
      mr = fmaxf(mr, red[i]);
      mc = fmaxf(mc, red[64 + i]);
    }
    red[128] = 1.f / (mr * mc);
  }
  __syncthreads();
  const float invden = red[128];
  for (int i = tid; i < 4096; i += 256)
    Zs[i] = Ag[(i & 63) * 64 + (i >> 6)] * invden;  // z = a^T/(||a||1 ||a||inf)
  __syncthreads();
#pragma unroll 1
  for (int it = 0; it < 6; ++it) {
    matmul64_f32<0>(T1, Ag, Zs, 1.0f, tid);    // az
    matmul64_f32<7>(T2, T1, T1, 1.0f, tid);    // az(7I - az)
    matmul64_f32<15>(T2, T1, T2, 1.0f, tid);   // az(15I - prev)
    matmul64_f32<13>(Zs, Zs, T2, 0.25f, tid);  // 0.25 z(13I - prev)
  }
  // write Z^T in f16 ([n][k] so downstream B-staging is contiguous)
  for (int i = tid; i < 4096; i += 256)
    ZT[(size_t)bh * 4096 + i] = (_Float16)Zs[(i & 63) * 64 + (i >> 6)];
}

// ===========================================================================
// w1 = a1 @ pinv(a2) : [BH][4096 x 64] f16.  grid = (64, 32)
// ===========================================================================
__global__ __launch_bounds__(256) void a1z_wmma(
    const _Float16* __restrict__ A1S, const _Float16* __restrict__ ZT,
    _Float16* __restrict__ W1) {
  __shared__ __align__(16) _Float16 As[64 * LDA];
  __shared__ __align__(16) _Float16 Bs[64 * LDA];
  const int tid = threadIdx.x, wave = tid >> 5, lane = tid & 31;
  const int m_sub = wave & 3, n_sub0 = (wave >> 2) << 1;
  const int row0 = blockIdx.x * 64;
  const int bh = blockIdx.y;
  const int sm = tid >> 2, skk = (tid & 3) * 8;
  v8f acc0 = {}, acc1 = {};
  for (int k0 = 0; k0 < 64; k0 += 32) {
    __syncthreads();
    stage_b128(As + sm * LDA + skk,
               A1S + ((size_t)bh * 4096 + row0 + sm) * 64 + k0 + skk);
    stage_b128(Bs + sm * LDA + skk,
               ZT + (size_t)bh * 4096 + sm * 64 + k0 + skk);
    stage_wait();
    __syncthreads();
    gemm_step(As, Bs, m_sub, n_sub0, lane, acc0, acc1);
  }
  const int gi = lane >> 4, l15 = lane & 15;
#pragma unroll
  for (int r = 0; r < 8; ++r) {
    const int tk = row0 + m_sub * 16 + r + 8 * gi;
#pragma unroll
    for (int t = 0; t < 2; ++t) {
      const int d = (n_sub0 + t) * 16 + l15;
      W1[((size_t)bh * 4096 + tk) * 64 + d] = (_Float16)(t ? acc1[r] : acc0[r]);
    }
  }
}

// ===========================================================================
// out = w1 @ a3v -> OATT[B][N][H*DH] f16.  grid = (64, 32)
// ===========================================================================
__global__ __launch_bounds__(256) void outk_wmma(
    const _Float16* __restrict__ W1, const _Float16* __restrict__ A3VT,
    _Float16* __restrict__ OATT) {
  __shared__ __align__(16) _Float16 As[64 * LDA];
  __shared__ __align__(16) _Float16 Bs[64 * LDA];
  const int tid = threadIdx.x, wave = tid >> 5, lane = tid & 31;
  const int m_sub = wave & 3, n_sub0 = (wave >> 2) << 1;
  const int row0 = blockIdx.x * 64;
  const int bh = blockIdx.y;
  const int b = bh >> 3, h = bh & 7;
  const int sm = tid >> 2, skk = (tid & 3) * 8;
  v8f acc0 = {}, acc1 = {};
  for (int k0 = 0; k0 < 64; k0 += 32) {
    __syncthreads();
    stage_b128(As + sm * LDA + skk,
               W1 + ((size_t)bh * 4096 + row0 + sm) * 64 + k0 + skk);
    stage_b128(Bs + sm * LDA + skk,
               A3VT + ((size_t)bh * 64 + sm) * 64 + k0 + skk);
    stage_wait();
    __syncthreads();
    gemm_step(As, Bs, m_sub, n_sub0, lane, acc0, acc1);
  }
  const int gi = lane >> 4, l15 = lane & 15;
#pragma unroll
  for (int r = 0; r < 8; ++r) {
    const int tk = row0 + m_sub * 16 + r + 8 * gi;
#pragma unroll
    for (int t = 0; t < 2; ++t) {
      const int d = (n_sub0 + t) * 16 + l15;
      OATT[((size_t)(b * 4096 + tk)) * 512 + h * 64 + d] =
          (_Float16)(t ? acc1[r] : acc0[r]);
    }
  }
}

// residual depthwise conv (33 taps over token dim of V), added into OATT
__global__ __launch_bounds__(256) void res_kernel(
    const _Float16* __restrict__ VT, const float* __restrict__ rw,
    _Float16* __restrict__ OATT) {
  const int idx = blockIdx.x * 256 + threadIdx.x;  // 32*4096*64
  const int bh = idx >> 18, rem = idx & 262143;
  const int tk = rem >> 6, d = rem & 63;
  const int b = bh >> 3, h = bh & 7;
  const _Float16* vrow = VT + ((size_t)bh * 64 + d) * 4096;  // contiguous taps
  float acc = 0.f;
#pragma unroll 1
  for (int j = 0; j < 33; ++j) {
    const int tn = tk + j - 16;
    const int ct = clampi(tn, 0, 4095);
    const float v = (float)vrow[ct];
    acc += (tn == ct) ? rw[h * 33 + j] * v : 0.f;
  }
  const size_t oi = ((size_t)(b * 4096 + tk)) * 512 + h * 64 + d;
  OATT[oi] = (_Float16)((float)OATT[oi] + acc);
}

// ===========================================================================
// projection: OPRJ[16384 x 800] = OATT[16384 x 512] @ out_w + out_b (f32 out)
// grid = (256, 13)
// ===========================================================================
__global__ __launch_bounds__(256) void proj_wmma(
    const _Float16* __restrict__ OATT, const _Float16* __restrict__ OWT,
    const float* __restrict__ out_b, float* __restrict__ OPRJ) {
  __shared__ __align__(16) _Float16 As[64 * LDA];
  __shared__ __align__(16) _Float16 Bs[64 * LDA];
  const int tid = threadIdx.x, wave = tid >> 5, lane = tid & 31;
  const int m_sub = wave & 3, n_sub0 = (wave >> 2) << 1;
  const int row0 = blockIdx.x * 64;
  const int col0 = blockIdx.y * 64;
  const int sm = tid >> 2, skk = (tid & 3) * 8;
  const bool bcol_ok = (col0 + sm) < 800;
  if (!bcol_ok) {  // zero masked B rows once
    v8h z;
#pragma unroll
    for (int e = 0; e < 8; ++e) z[e] = (_Float16)0;
    *(v8h*)(Bs + sm * LDA + skk) = z;
  }
  v8f acc0 = {}, acc1 = {};
  for (int k0 = 0; k0 < 512; k0 += 32) {
    __syncthreads();
    stage_b128(As + sm * LDA + skk,
               OATT + (size_t)(row0 + sm) * 512 + k0 + skk);
    if (bcol_ok)
      stage_b128(Bs + sm * LDA + skk,
                 OWT + (size_t)(col0 + sm) * 512 + k0 + skk);
    stage_wait();
    __syncthreads();
    gemm_step(As, Bs, m_sub, n_sub0, lane, acc0, acc1);
  }
  const int gi = lane >> 4, l15 = lane & 15;
#pragma unroll
  for (int t = 0; t < 2; ++t) {
    const int col = col0 + (n_sub0 + t) * 16 + l15;
    if (col < 800) {
      const float bia = out_b[col];
#pragma unroll
      for (int r = 0; r < 8; ++r) {
        const int row = row0 + m_sub * 16 + r + 8 * gi;
        OPRJ[(size_t)row * 800 + col] = (t ? acc1[r] : acc0[r]) + bia;
      }
    }
  }
}

// ===========================================================================
// LayerNorm(800) + head (800 -> 16) + sigmoid.  one block per row (16384)
// ===========================================================================
__global__ __launch_bounds__(256) void ln_head_kernel(
    const float* __restrict__ OPRJ, const float* __restrict__ ln_g,
    const float* __restrict__ ln_b, const float* __restrict__ head_w,
    const float* __restrict__ head_b, float* __restrict__ out) {
  __shared__ float rowbuf[800];
  __shared__ float sbuf[256];
  __shared__ float stats[2];
  const int tid = threadIdx.x;
  const size_t row = blockIdx.x;
  float s1 = 0.f, s2 = 0.f;
  for (int e = tid; e < 800; e += 256) {
    const float v = OPRJ[row * 800 + e];
    rowbuf[e] = v;
    s1 += v;
    s2 += v * v;
  }
  sbuf[tid] = s1;
  __syncthreads();
  if (tid == 0) {
    float t = 0.f;
    for (int i = 0; i < 256; ++i) t += sbuf[i];
    stats[0] = t * (1.f / 800.f);
  }
  __syncthreads();
  sbuf[tid] = s2;
  __syncthreads();
  if (tid == 0) {
    float t = 0.f;
    for (int i = 0; i < 256; ++i) t += sbuf[i];
    const float mean = stats[0];
    stats[1] = rsqrtf(t * (1.f / 800.f) - mean * mean + 1e-5f);
  }
  __syncthreads();
  const float mean = stats[0], rstd = stats[1];
  for (int e = tid; e < 800; e += 256)
    rowbuf[e] = (rowbuf[e] - mean) * rstd * ln_g[e] + ln_b[e];
  __syncthreads();
  const int o = tid >> 4, li = tid & 15;
  float p = 0.f;
  for (int e = li; e < 800; e += 16) p += rowbuf[e] * head_w[e * 16 + o];
  sbuf[tid] = p;
  __syncthreads();
  if (li == 0) {
    float s = 0.f;
    for (int j = 0; j < 16; ++j) s += sbuf[o * 16 + j];
    s += head_b[o];
    out[row * 16 + o] = 1.f / (1.f + __expf(-s));
  }
}

// ===========================================================================
// host-side launch
// ===========================================================================
static constexpr size_t ALN(size_t x) { return (x + 255) & ~(size_t)255; }

extern "C" void kernel_launch(void* const* d_in, const int* in_sizes, int n_in,
                              void* d_out, int out_size, void* d_ws,
                              size_t ws_size, hipStream_t stream) {
  (void)in_sizes; (void)n_in; (void)out_size; (void)ws_size;
  const float* x = (const float*)d_in[0];
  const float *w0 = (const float*)d_in[1],  *b0 = (const float*)d_in[2],
              *g0 = (const float*)d_in[3],  *be0 = (const float*)d_in[4],
              *mu0 = (const float*)d_in[5], *var0 = (const float*)d_in[6];
  const float *w1 = (const float*)d_in[7],  *b1 = (const float*)d_in[8],
              *g1 = (const float*)d_in[9],  *be1 = (const float*)d_in[10],
              *mu1 = (const float*)d_in[11],*var1 = (const float*)d_in[12];
  const float *w2 = (const float*)d_in[13], *b2 = (const float*)d_in[14],
              *g2 = (const float*)d_in[15], *be2 = (const float*)d_in[16],
              *mu2 = (const float*)d_in[17],*var2 = (const float*)d_in[18];
  const float *w3 = (const float*)d_in[19], *b3 = (const float*)d_in[20],
              *g3 = (const float*)d_in[21], *be3 = (const float*)d_in[22],
              *mu3 = (const float*)d_in[23],*var3 = (const float*)d_in[24];
  const float* qkv_w = (const float*)d_in[25];
  const float* out_w = (const float*)d_in[26];
  const float* out_b = (const float*)d_in[27];
  const float* res_w = (const float*)d_in[28];
  const float* ln_g = (const float*)d_in[29];
  const float* ln_b = (const float*)d_in[30];
  const float* head_w = (const float*)d_in[31];
  const float* head_b = (const float*)d_in[32];
  float* out = (float*)d_out;

  char* ws = (char*)d_ws;
  size_t off = 0;
  auto take = [&](size_t bytes) {
    char* p = ws + off;
    off += ALN(bytes);
    return p;
  };
  _Float16* A0   = (_Float16*)take((size_t)4 * 64 * 65536 * 2);
  _Float16* A1   = (_Float16*)take((size_t)4 * 128 * 65536 * 2);
  _Float16* A2   = (_Float16*)take((size_t)4 * 256 * 65536 * 2);
  _Float16* TOK  = (_Float16*)take((size_t)4 * 4096 * 800 * 2);
  _Float16* Qb   = (_Float16*)take((size_t)32 * 4096 * 64 * 2);
  _Float16* Kb   = (_Float16*)take((size_t)32 * 4096 * 64 * 2);
  _Float16* VT   = (_Float16*)take((size_t)32 * 64 * 4096 * 2);
  _Float16* QL   = (_Float16*)take((size_t)32 * 64 * 64 * 2);
  _Float16* KL   = (_Float16*)take((size_t)32 * 64 * 64 * 2);
  _Float16* A1S  = (_Float16*)take((size_t)32 * 4096 * 64 * 2);
  float*    A2S  = (float*)take((size_t)32 * 64 * 64 * 4);
  _Float16* ZT   = (_Float16*)take((size_t)32 * 64 * 64 * 2);
  _Float16* S3E  = (_Float16*)take((size_t)32 * 64 * 4096 * 2);
  float*    RS3  = (float*)take((size_t)32 * 64 * 4);
  _Float16* A3VT = (_Float16*)take((size_t)32 * 64 * 64 * 2);
  _Float16* W1b  = (_Float16*)take((size_t)32 * 4096 * 64 * 2);
  _Float16* OATT = (_Float16*)take((size_t)4 * 4096 * 512 * 2);
  float*    OPRJ = (float*)take((size_t)4 * 4096 * 800 * 4);
  _Float16* WC1  = (_Float16*)take((size_t)128 * 576 * 2);
  _Float16* WC2  = (_Float16*)take((size_t)256 * 1152 * 2);
  _Float16* WC3  = (_Float16*)take((size_t)800 * 4096 * 2);
  _Float16* WQKVT= (_Float16*)take((size_t)1536 * 800 * 2);
  _Float16* OWT  = (_Float16*)take((size_t)800 * 512 * 2);

  // ---- weight prep (f16 conversions / transposes) ----
  cvt_f16_kernel<<<288, 256, 0, stream>>>(w1, WC1, 128 * 576);
  cvt_f16_kernel<<<1152, 256, 0, stream>>>(w2, WC2, 256 * 1152);
  cvt_f16_kernel<<<12800, 256, 0, stream>>>(w3, WC3, 800 * 4096);
  transpose_f16_kernel<<<4800, 256, 0, stream>>>(qkv_w, WQKVT, 800, 1536);
  transpose_f16_kernel<<<1600, 256, 0, stream>>>(out_w, OWT, 512, 800);

  // ---- patch-embed conv stack ----
  conv0_kernel<<<65536, 256, 0, stream>>>(x, w0, b0, g0, be0, mu0, var0, A0);
  conv3x3_wmma<64, 128><<<dim3(1024, 2, 4), 256, 0, stream>>>(
      A0, WC1, b1, g1, be1, mu1, var1, A1);
  conv3x3_wmma<128, 256><<<dim3(1024, 4, 4), 256, 0, stream>>>(
      A1, WC2, b2, g2, be2, mu2, var2, A2);
  conv4x4_wmma<<<dim3(64, 13, 4), 256, 0, stream>>>(A2, WC3, b3, g3, be3, mu3,
                                                    var3, TOK);
  // ---- qkv + landmarks ----
  qkv_wmma<<<dim3(256, 24), 256, 0, stream>>>(TOK, WQKVT, Qb, Kb, VT);
  landmark_kernel<<<512, 256, 0, stream>>>(Qb, QL);
  landmark_kernel<<<512, 256, 0, stream>>>(Kb, KL);
  // ---- nystrom attention ----
  attn_small_wmma<false><<<dim3(64, 32), 256, 0, stream>>>(Qb, KL, A1S, nullptr);
  attn_small_wmma<true><<<dim3(1, 32), 256, 0, stream>>>(QL, KL, nullptr, A2S);
  (void)hipMemsetAsync(RS3, 0, (size_t)32 * 64 * 4, stream);
  s3_wmma<<<dim3(64, 32), 256, 0, stream>>>(QL, Kb, S3E, RS3);
  a3v_wmma<<<32, 256, 0, stream>>>(S3E, VT, RS3, A3VT);
  pinv_kernel<<<32, 256, 0, stream>>>(A2S, ZT);
  a1z_wmma<<<dim3(64, 32), 256, 0, stream>>>(A1S, ZT, W1b);
  outk_wmma<<<dim3(64, 32), 256, 0, stream>>>(W1b, A3VT, OATT);
  res_kernel<<<32768, 256, 0, stream>>>(VT, res_w, OATT);
  // ---- projection + LN + head ----
  proj_wmma<<<dim3(256, 13), 256, 0, stream>>>(OATT, OWT, out_b, OPRJ);
  ln_head_kernel<<<16384, 256, 0, stream>>>(OPRJ, ln_g, ln_b, head_w, head_b,
                                            out);
}
